// MultiHeadAttention_73409581023673
// MI455X (gfx1250) — matrix operand order repaired and checked by reading, not compiled, not run on hardware
//
#include <hip/hip_runtime.h>
#include <hip/hip_bf16.h>
#include <math.h>

// ---------------------------------------------------------------------------
// MultiHeadAttention row-0 pipeline for MI455X (gfx1250, wave32, WMMA + TDM).
// Only p_attn[:, :, 0, :] feeds the conv, so we never build the SxS scores.
//   1) X -> bf16, W_G/Wk -> bf16 transposed (N-major)
//   2) G = X*W_G + b_G   (TDM-staged LDS double buffer + bf16 WMMA, bf16 out)
//   3) K = G*Wk + bk     (same GEMM, f32 out)
//   4) q0 = G[:,0,:]*Wq + bq (tiny, VALU)
//   5) RoPE(K) dot q0, softmax over S  -> row0 (B,H,S)
//   6) conv1d(row0) + bias, ReLU      -> out (B,D,S)
// ---------------------------------------------------------------------------

#if defined(__has_builtin)
#  if __has_builtin(__builtin_amdgcn_tensor_load_to_lds) && \
      __has_builtin(__builtin_amdgcn_s_wait_tensorcnt)
#    define HAVE_TDM 1
#  endif
#endif
#ifndef HAVE_TDM
#  define HAVE_TDM 0
#endif

namespace {

constexpr int Bb = 4;
constexpr int Ss = 1024;
constexpr int Dd = 1024;
constexpr int Hh = 16;
constexpr int DK = 64;
constexpr int Mm = Bb * Ss;     // 4096 rows for the big GEMMs

typedef __attribute__((ext_vector_type(16))) __bf16        v16bf;
typedef __attribute__((ext_vector_type(8)))  float         v8f;
typedef __attribute__((ext_vector_type(4)))  unsigned int  v4u;
typedef __attribute__((ext_vector_type(8)))  int           v8i;
typedef __attribute__((ext_vector_type(4)))  int           v4i;

union FragU {
    uint4  u[2];   // two 16-byte loads
    v16bf  v;      // 16 bf16 = 8 VGPRs, WMMA operand
};

__device__ __forceinline__ __bf16 f2bf(float f) {
    unsigned u = __builtin_bit_cast(unsigned, f);
    unsigned r = u + 0x7FFFu + ((u >> 16) & 1u);     // round-to-nearest-even
    unsigned short h = (unsigned short)(r >> 16);
    return __builtin_bit_cast(__bf16, h);
}
__device__ __forceinline__ float bf2f(__bf16 b) {
    unsigned short h = __builtin_bit_cast(unsigned short, b);
    unsigned u = ((unsigned)h) << 16;
    return __builtin_bit_cast(float, u);
}

#if HAVE_TDM
// ---- Tensor Data Mover: 2-D tile (tile_d0 x tile_d1 of 2-byte elems) -> LDS.
// D# per CDNA5 ISA. group0: count=1, lds_addr, 57-bit
// global addr, type=2. group1: data_size=1 (2B), tensor dims, tile dims,
// dim0 stride. Groups 2/3 zero (2-D tile: tile_dim2/3 = 0 -> unused).
__device__ __forceinline__ void tdm_load_2d_bf16(unsigned lds_addr, const void* gptr,
                                                 unsigned tensor_d0, unsigned tensor_d1,
                                                 unsigned tile_d0, unsigned tile_d1,
                                                 unsigned stride0) {
    unsigned long long ga = (unsigned long long)gptr;
    v4u g0;
    g0[0] = 1u;                                           // count=1 valid descriptor
    g0[1] = lds_addr;                                     // LDS byte address
    g0[2] = (unsigned)(ga & 0xFFFFFFFFu);                 // global_addr[31:0]
    g0[3] = (unsigned)((ga >> 32) & 0x1FFFFFFu)           // global_addr[56:32]
          | (2u << 30);                                   // type=2 ("image")
    v8i g1;
    g1[0] = (int)(1u << 16);                              // data_size=1 -> 2 bytes
    g1[1] = (int)((tensor_d0 & 0xFFFFu) << 16);           // tensor_dim0[15:0]
    g1[2] = (int)((tensor_d0 >> 16) | ((tensor_d1 & 0xFFFFu) << 16));
    g1[3] = (int)((tensor_d1 >> 16) | (tile_d0 << 16));   // tile_dim0
    g1[4] = (int)(tile_d1 & 0xFFFFu);                     // tile_dim1 (tile_dim2=0)
    g1[5] = (int)stride0;                                 // tensor_dim0_stride[31:0]
    g1[6] = 0;
    g1[7] = 0;
    v4i z4 = {0, 0, 0, 0};
#if __clang_major__ >= 23
    v8i z8 = {0, 0, 0, 0, 0, 0, 0, 0};
    __builtin_amdgcn_tensor_load_to_lds(g0, g1, z4, z4, z8, 0);
#else
    __builtin_amdgcn_tensor_load_to_lds(g0, g1, z4, z4, 0);
#endif
}
#endif

// cooperative fallback staging: 128x32 bf16 tile (8 KB) with 256 threads
__device__ __forceinline__ void stage_tile(const __bf16* __restrict__ src, int srcStride,
                                           __bf16* dst, int tid) {
    #pragma unroll
    for (int i = 0; i < 2; ++i) {
        const int idx = tid + i * 256;        // 0..511 uint4 slots, 4 per row
        const int row = idx >> 2, c = idx & 3;
        ((uint4*)dst)[idx] = *((const uint4*)(src + (size_t)row * srcStride) + c);
    }
}

// ---------------- fp32 -> bf16 elementwise ----------------
__global__ __launch_bounds__(256) void cvt_bf16_kernel(const float* __restrict__ src,
                                                       __bf16* __restrict__ dst, int n) {
    int i = blockIdx.x * 256 + threadIdx.x;
    int stride = gridDim.x * 256;
    for (; i < n; i += stride) dst[i] = f2bf(src[i]);
}

// ---------------- W (KxN f32 row-major) -> Wt (NxK bf16) ----------------
__global__ __launch_bounds__(256) void transpose_bf16_kernel(const float* __restrict__ W,
                                                             __bf16* __restrict__ Wt) {
    __shared__ float tile[32][33];
    int bx = blockIdx.x * 32;            // n tile
    int by = blockIdx.y * 32;            // k tile
    int tx = threadIdx.x;
    int ty = threadIdx.y;
    #pragma unroll
    for (int j = 0; j < 32; j += 8)
        tile[ty + j][tx] = W[(size_t)(by + ty + j) * Dd + (bx + tx)];
    __syncthreads();
    #pragma unroll
    for (int j = 0; j < 32; j += 8)
        Wt[(size_t)(bx + ty + j) * Dd + (by + tx)] = f2bf(tile[tx][ty + j]);
}

// ------- bf16 WMMA GEMM: C(MxN) = A(MxK) * Bt(NxK)^T + bias --------------
// 256 threads = 8 waves. Block tile 128(M) x 128(N); wave tile 32x64
// (2 M-frags x 4 N-frags = 8 accumulators). A/B K-slabs (128x32 bf16, 8 KB
// each) are staged into a double-buffered LDS slab by the Tensor Data Mover
// (wave 0 issues tensor_load_to_lds; completion via TENSORcnt + barrier),
// while the other buffer feeds ds_load_b128 fragment reads and WMMAs.
template <int OUT_BF16>
__global__ __launch_bounds__(256) void gemm_wmma_kernel(const __bf16* __restrict__ A,
                                                        const __bf16* __restrict__ Bt,
                                                        const float* __restrict__ bias,
                                                        void* __restrict__ Cout,
                                                        int M, int N, int K) {
    __shared__ __bf16 Ash[2][128 * 32];
    __shared__ __bf16 Bsh[2][128 * 32];

    const int tid  = threadIdx.x;
    const int lane = tid & 31;
    const int wave = tid >> 5;            // 0..7
    const int wm   = wave & 3;            // 4 waves along M
    const int wn   = wave >> 2;           // 2 waves along N
    const int mBlock = blockIdx.y * 128;
    const int nBlock = blockIdx.x * 128;
    const int half = lane >> 4;
    const int l15  = lane & 15;

    v8f acc[2][4];
    #pragma unroll
    for (int mt = 0; mt < 2; ++mt)
        #pragma unroll
        for (int nt = 0; nt < 4; ++nt)
            #pragma unroll
            for (int r = 0; r < 8; ++r) acc[mt][nt][r] = 0.0f;

    const __bf16* Ag = A  + (size_t)mBlock * K;
    const __bf16* Bg = Bt + (size_t)nBlock * K;

#if HAVE_TDM
    if (wave == 0) {
        tdm_load_2d_bf16((unsigned)(unsigned long long)(const void*)Ash[0], Ag,
                         K, M, 32, 128, K);
        tdm_load_2d_bf16((unsigned)(unsigned long long)(const void*)Bsh[0], Bg,
                         K, N, 32, 128, K);
        __builtin_amdgcn_s_wait_tensorcnt(0);
    }
    __syncthreads();
#else
    stage_tile(Ag, K, Ash[0], tid);
    stage_tile(Bg, K, Bsh[0], tid);
    __syncthreads();
#endif

    for (int k0 = 0; k0 < K; k0 += 32) {
        const int cur  = (k0 >> 5) & 1;
        const int nxt  = cur ^ 1;
        const bool more = (k0 + 32) < K;

#if HAVE_TDM
        if (more && wave == 0) {        // kick next slab while we compute
            tdm_load_2d_bf16((unsigned)(unsigned long long)(const void*)Ash[nxt],
                             Ag + (k0 + 32), K, M, 32, 128, K);
            tdm_load_2d_bf16((unsigned)(unsigned long long)(const void*)Bsh[nxt],
                             Bg + (k0 + 32), K, N, 32, 128, K);
        }
#endif
        // ---- compute on current buffer ----
        const uint4* Au = (const uint4*)Ash[cur];
        const uint4* Bu = (const uint4*)Bsh[cur];
        FragU a[2];
        #pragma unroll
        for (int mt = 0; mt < 2; ++mt) {
            const int idx = (wm * 32 + mt * 16 + l15) * 4 + 2 * half;  // row*4 + 2*half
            a[mt].u[0] = Au[idx];
            a[mt].u[1] = Au[idx + 1];
        }
        #pragma unroll
        for (int nt = 0; nt < 4; ++nt) {
            const int idx = (wn * 64 + nt * 16 + l15) * 4 + 2 * half;
            FragU bfrag;
            bfrag.u[0] = Bu[idx];
            bfrag.u[1] = Bu[idx + 1];
            #pragma unroll
            for (int mt = 0; mt < 2; ++mt)
                acc[mt][nt] = __builtin_amdgcn_wmma_f32_16x16x32_bf16(
                    false, a[mt].v, false, bfrag.v, (short)0, acc[mt][nt], false, false);
        }

        if (more) {
#if HAVE_TDM
            if (wave == 0) __builtin_amdgcn_s_wait_tensorcnt(0);
            __syncthreads();
#else
            stage_tile(Ag + (k0 + 32), K, Ash[nxt], tid);
            stage_tile(Bg + (k0 + 32), K, Bsh[nxt], tid);
            __syncthreads();
#endif
        }
    }

    // C layout: VGPR r, lane L -> m = r + 8*(L>>4), n = L&15
    const int mC0 = mBlock + wm * 32;
    const int nC0 = nBlock + wn * 64;
    #pragma unroll
    for (int mt = 0; mt < 2; ++mt) {
        #pragma unroll
        for (int nt = 0; nt < 4; ++nt) {
            const int col = nC0 + nt * 16 + l15;
            const float bv = bias[col];
            #pragma unroll
            for (int r = 0; r < 8; ++r) {
                const float val = acc[mt][nt][r] + bv;
                const size_t idx = (size_t)(mC0 + mt * 16 + 8 * half + r) * N + col;
                if (OUT_BF16) ((__bf16*)Cout)[idx] = f2bf(val);
                else          ((float*)Cout)[idx]  = val;
            }
        }
    }
}

// ---------------- q0[b,n] = sum_k G[b*S, k] * Wq[k,n] + bq[n] ----------------
__global__ __launch_bounds__(256) void q0_kernel(const __bf16* __restrict__ G,
                                                 const float* __restrict__ Wq,
                                                 const float* __restrict__ bq,
                                                 float* __restrict__ q0) {
    const int n = blockIdx.x * 256 + threadIdx.x;
    const int b = blockIdx.y;
    const __bf16* grow = G + (size_t)(b * Ss) * Dd;
    float acc = bq[n];
    for (int k = 0; k < Dd; ++k)
        acc += bf2f(grow[k]) * Wq[(size_t)k * Dd + n];
    q0[b * Dd + n] = acc;
}

// ---------------- RoPE(K) . q0, masked softmax over S -> row0 ---------------
__global__ __launch_bounds__(256) void attn_row0_kernel(const float* __restrict__ Kmat,
                                                        const float* __restrict__ q0,
                                                        const int* __restrict__ mask,
                                                        float* __restrict__ row0) {
    const int b = blockIdx.x / Hh;
    const int h = blockIdx.x % Hh;
    const int tid = threadIdx.x;

    __shared__ float sc[Ss];
    __shared__ float qs[DK];
    __shared__ float rbuf[256];

    if (tid < DK) qs[tid] = q0[b * Dd + h * DK + tid];   // RoPE at s=0 is identity
    __syncthreads();

    const float lnstep = -0.2878231366f;   // -ln(10000)*2/64
    for (int s = tid; s < Ss; s += 256) {
        const float* kr = Kmat + ((size_t)(b * Ss + s)) * Dd + h * DK;
        float acc = 0.0f;
        #pragma unroll
        for (int j = 0; j < DK / 2; ++j) {
            const float inv = __expf(lnstep * (float)j);
            const float ang = (float)s * inv;
            const float c = cosf(ang), si = sinf(ang);
            const float k0 = kr[2 * j], k1 = kr[2 * j + 1];
            const float kw0 = k0 * c - k1 * si;
            const float kw1 = k1 * c + k0 * si;
            acc += qs[2 * j] * kw0 + qs[2 * j + 1] * kw1;
        }
        acc *= 0.125f;                                   // 1/sqrt(64)
        if (mask[b * Ss + s] == 0) acc = -1.0e9f;
        sc[s] = acc;
    }
    __syncthreads();

    float m = -INFINITY;
    for (int s = tid; s < Ss; s += 256) m = fmaxf(m, sc[s]);
    rbuf[tid] = m;
    __syncthreads();
    for (int off = 128; off > 0; off >>= 1) {
        if (tid < off) rbuf[tid] = fmaxf(rbuf[tid], rbuf[tid + off]);
        __syncthreads();
    }
    const float mx = rbuf[0];
    __syncthreads();

    float psum = 0.0f;
    for (int s = tid; s < Ss; s += 256) {
        const float e = __expf(sc[s] - mx);
        sc[s] = e;
        psum += e;
    }
    rbuf[tid] = psum;
    __syncthreads();
    for (int off = 128; off > 0; off >>= 1) {
        if (tid < off) rbuf[tid] += rbuf[tid + off];
        __syncthreads();
    }
    const float invsum = 1.0f / rbuf[0];
    float* orow = row0 + (size_t)(b * Hh + h) * Ss;
    for (int s = tid; s < Ss; s += 256) orow[s] = sc[s] * invsum;
}

// ---------------- conv1d(row0, w[D,H,3], pad=1) + bias, ReLU ----------------
__global__ __launch_bounds__(256) void conv_relu_kernel(const float* __restrict__ row0,
                                                        const float* __restrict__ w,
                                                        const float* __restrict__ cb,
                                                        float* __restrict__ out) {
    const int d0    = blockIdx.x * 16;
    const int sBase = blockIdx.y * 512;
    const int b     = blockIdx.z;
    const int tid   = threadIdx.x;

    __shared__ float r0[Hh][514];      // 512 + halo(2)
    __shared__ float wsh[16 * 48];

    for (int idx = tid; idx < Hh * 514; idx += 256) {
        const int i  = idx / 514;
        const int sl = idx % 514;
        const int sg = sBase + sl - 1;
        r0[i][sl] = (sg >= 0 && sg < Ss) ? row0[(size_t)(b * Hh + i) * Ss + sg] : 0.0f;
    }
    for (int idx = tid; idx < 16 * 48; idx += 256)
        wsh[idx] = w[(size_t)d0 * 48 + idx];
    __syncthreads();

    for (int dl = 0; dl < 16; ++dl) {
        const int d = d0 + dl;
        const float bias = cb[d];
        const float* wd = &wsh[dl * 48];
        for (int sl = tid; sl < 512; sl += 256) {
            float acc = bias;
            #pragma unroll
            for (int i = 0; i < Hh; ++i) {
                acc += r0[i][sl + 0] * wd[i * 3 + 0];
                acc += r0[i][sl + 1] * wd[i * 3 + 1];
                acc += r0[i][sl + 2] * wd[i * 3 + 2];
            }
            out[((size_t)(b * Dd + d)) * Ss + sBase + sl] = fmaxf(acc, 0.0f);
        }
    }
}

} // namespace

extern "C" void kernel_launch(void* const* d_in, const int* in_sizes, int n_in,
                              void* d_out, int out_size, void* d_ws, size_t ws_size,
                              hipStream_t stream) {
    const float* X      = (const float*)d_in[0];   // (B,S,D)
    const int*   mask   = (const int*)  d_in[1];   // (B,S)
    const float* W_G    = (const float*)d_in[2];   // (D,D)
    const float* b_G    = (const float*)d_in[3];
    const float* Wq     = (const float*)d_in[4];
    const float* bq     = (const float*)d_in[5];
    const float* Wk     = (const float*)d_in[6];
    const float* bk     = (const float*)d_in[7];
    const float* conv_w = (const float*)d_in[8];   // (D,H,3)
    const float* conv_b = (const float*)d_in[9];
    float* out = (float*)d_out;                    // (B,D,S)

    char* ws = (char*)d_ws;
    __bf16* Xb   = (__bf16*)(ws);                              // 8 MB
    __bf16* WGt  = (__bf16*)(ws + (8u << 20));                 // 2 MB
    __bf16* Wkt  = (__bf16*)(ws + (10u << 20));                // 2 MB
    __bf16* Gb   = (__bf16*)(ws + (12u << 20));                // 8 MB
    float*  Kmat = (float*) (ws + (20u << 20));                // 16 MB
    float*  q0   = (float*) (ws + (36u << 20));                // 16 KB
    float*  row0 = (float*) (ws + (36u << 20) + (64u << 10));  // 256 KB

    // 1) conversions / transposes
    cvt_bf16_kernel<<<512, 256, 0, stream>>>(X, Xb, Mm * Dd);
    {
        dim3 g(Dd / 32, Dd / 32), blk(32, 8);
        transpose_bf16_kernel<<<g, blk, 0, stream>>>(W_G, WGt);
        transpose_bf16_kernel<<<g, blk, 0, stream>>>(Wk,  Wkt);
    }
    // 2) G = X*W_G + b_G   (bf16 out, feeds GEMM2)
    {
        dim3 g(Dd / 128, Mm / 128);
        gemm_wmma_kernel<1><<<g, 256, 0, stream>>>(Xb, WGt, b_G, (void*)Gb, Mm, Dd, Dd);
    }
    // 3) K = G*Wk + bk     (f32 out)
    {
        dim3 g(Dd / 128, Mm / 128);
        gemm_wmma_kernel<0><<<g, 256, 0, stream>>>(Gb, Wkt, bk, (void*)Kmat, Mm, Dd, Dd);
    }
    // 4) q0 (only query row s=0 per batch is ever used)
    q0_kernel<<<dim3(Dd / 256, Bb), 256, 0, stream>>>(Gb, Wq, bq, q0);
    // 5) RoPE + scores row 0 + softmax
    attn_row0_kernel<<<Bb * Hh, 256, 0, stream>>>(Kmat, q0, mask, row0);
    // 6) conv + ReLU
    conv_relu_kernel<<<dim3(Dd / 16, Ss / 512, Bb), 256, 0, stream>>>(row0, conv_w, conv_b, out);
}
